// MultiHeadAttentionSparse_309237646006
// MI455X (gfx1250) — matrix operand order repaired and checked by reading, not compiled, not run on hardware
//
#include <hip/hip_runtime.h>
#include <hip/hip_bf16.h>
#include <math.h>

// ---------------- problem constants ----------------
#define EMB   1024
#define HEADS 16
#define HDIM  64
#define SEQ   2048
#define BATCH 2
#define WIN   64
#define NTOK  (BATCH * SEQ)      // 4096
#define NEGF  (-3.0e38f)

typedef __attribute__((ext_vector_type(16))) _Float16 v16h;
typedef __attribute__((ext_vector_type(8)))  _Float16 h8;
typedef __attribute__((ext_vector_type(8)))  float    v8f;
typedef __attribute__((ext_vector_type(4)))  float    f4;

// ---------------- WMMA wrapper ----------------
__device__ __forceinline__ v8f wmma_f16(v16h a, v16h b, v8f c) {
  // D = A(16x32 f16) * B(32x16 f16) + C(16x16 f32)
  return __builtin_amdgcn_wmma_f32_16x16x32_f16(
      /*neg_a=*/false, a, /*neg_b=*/false, b,
      /*c_mod=*/(short)0, c, /*reuse_a=*/false, /*reuse_b=*/false);
}

// ---------------- fragment loaders (ISA 7.12.2 layouts) ----------------
// A 16x32 f16: lane L<16 -> row L, K = {hi*8+0..7} in elems 0..7, {16+hi*8+0..7} in 8..15
// caller passes p = &A[row][k0 + hi*8]
__device__ __forceinline__ v16h load_a(const _Float16* p) {
  h8 lo = *(const h8*)(p);
  h8 hv = *(const h8*)(p + 16);
  v16h f;
#pragma unroll
  for (int i = 0; i < 8; ++i) { f[i] = lo[i]; f[i + 8] = hv[i]; }
  return f;
}
// B 32x16 f16: lane L<16 -> col L, K = {hi*8+0..7} in elems 0..7, {16+hi*8+0..7} in 8..15
// caller passes p = &Brow[col][k0 + hi*8]  (B stored as N x K row-major)
__device__ __forceinline__ v16h load_b(const _Float16* p) {
  h8 lo = *(const h8*)(p);
  h8 hv = *(const h8*)(p + 16);
  v16h f;
#pragma unroll
  for (int i = 0; i < 8; ++i) { f[i] = lo[i]; f[i + 8] = hv[i]; }
  return f;
}

// ---------------- fp32 -> fp16 pre-conversion (memory-bound, one pass) ----------------
// Keeps the GEMM K-loop free of v_cvt work so the WMMA pipe is the bottleneck.
__global__ void __launch_bounds__(256)
cvt_f16(const float* __restrict__ src, _Float16* __restrict__ dst, int n) {
  const int i = (blockIdx.x * 256 + threadIdx.x) * 8;
  if (i >= n) return;
  f4 a = *(const f4*)(src + i);
  f4 b = *(const f4*)(src + i + 4);
  h8 o;
#pragma unroll
  for (int j = 0; j < 4; ++j) {
    o[j]     = (_Float16)a[j];
    o[j + 4] = (_Float16)b[j];
  }
  *(h8*)(dst + i) = o;
}

// ---------------- projection / FC GEMM ----------------
// out[n,e] = sum_k A[n,k] * W[e,k] + bias[e]; wave tile 32(M) x 64(N)
// MODE: 0=Q (RoPE, [b,h,s,d] f16)  1=K (RoPE, [b,h,s,d] f16)
//       2=V ([b,h,d,s] f16 transposed)  3=FC (f32 row-major out)
#define MODE_Q 0
#define MODE_K 1
#define MODE_V 2
#define MODE_FC 3

template <int MODE>
__global__ void __launch_bounds__(256)
proj_gemm(const _Float16* __restrict__ A, const _Float16* __restrict__ W,
          const float* __restrict__ bias, void* __restrict__ dst) {
  const int lane = threadIdx.x & 31;
  const int wid  = threadIdx.x >> 5;
  const int r    = lane & 15;
  const int hi   = lane >> 4;
  const int rowBase = blockIdx.y * 64  + (wid >> 2) * 32;   // M
  const int colBase = blockIdx.x * 256 + (wid & 3)  * 64;   // N

  const _Float16* pa0 = A + (size_t)(rowBase + r) * EMB + hi * 8;
  const _Float16* pa1 = A + (size_t)(rowBase + 16 + r) * EMB + hi * 8;
  const _Float16* pw  = W + (size_t)(colBase + r) * EMB + hi * 8;

  v8f acc[2][4] = {};

  for (int k0 = 0; k0 < EMB; k0 += 32) {
    // pure loads + WMMA in the hot loop; prefetch next K-step (global_prefetch_b8)
    __builtin_prefetch(pa0 + k0 + 32, 0, 1);
    __builtin_prefetch(pw + k0 + 32, 0, 1);
    v16h af[2], bf[4];
    af[0] = load_a(pa0 + k0);
    af[1] = load_a(pa1 + k0);
#pragma unroll
    for (int ni = 0; ni < 4; ++ni)
      bf[ni] = load_b(pw + (size_t)ni * 16 * EMB + k0);
#pragma unroll
    for (int mi = 0; mi < 2; ++mi)
#pragma unroll
      for (int ni = 0; ni < 4; ++ni)
        acc[mi][ni] = wmma_f16(af[mi], bf[ni], acc[mi][ni]);
  }

  // epilogue: bias (+ RoPE for Q/K) + layout-specific store
#pragma unroll
  for (int mi = 0; mi < 2; ++mi) {
#pragma unroll
    for (int ni = 0; ni < 4; ++ni) {
#pragma unroll
      for (int v = 0; v < 8; ++v) {
        const int row = rowBase + mi * 16 + v + 8 * hi;   // token 0..4095
        const int col = colBase + ni * 16 + r;            // 0..1023
        float x = acc[mi][ni][v] + bias[col];
        if constexpr (MODE == MODE_FC) {
          ((float*)dst)[(size_t)row * EMB + col] = x;
        } else {
          const int b = row >> 11, s = row & (SEQ - 1);
          const int h = col >> 6,  d = col & (HDIM - 1);
          if constexpr (MODE == MODE_Q || MODE == MODE_K) {
            // interleaved RoPE: pair (2i, 2i+1) sits in adjacent lanes
            const int de = d & ~1;                         // 2i
            const float ang =
                (float)s * exp2f(-(float)de * (13.287712379549449f / 64.0f));
            const float c = cosf(ang), sn = sinf(ang);
            const float partner = __shfl_xor(x, 1, 32);
            x = (d & 1) ? fmaf(x, c,  partner * sn)
                        : fmaf(x, c, -partner * sn);
          }
          const _Float16 xh = (_Float16)x;
          if constexpr (MODE == MODE_V) {                  // V transposed [b,h,d,s]
            ((_Float16*)dst)[(((size_t)(b * HEADS + h) * HDIM + d) * SEQ) + s] = xh;
          } else {                                         // Q/K [b,h,s,d]
            ((_Float16*)dst)[(((size_t)(b * HEADS + h) * SEQ + s) * HDIM) + d] = xh;
          }
        }
      }
    }
  }
}

// ---------------- fused sparse flash attention ----------------
// grid = (S/64, B*H), block = 128 (4 waves); wave w owns query rows q0..q0+15
__global__ void __launch_bounds__(128)
sparse_attn(const _Float16* __restrict__ Qh, const _Float16* __restrict__ Kh,
            const _Float16* __restrict__ Vt, const float* __restrict__ rel_bias,
            const unsigned char* __restrict__ mask,
            const float* __restrict__ pos_coeff, _Float16* __restrict__ attn_out) {
  __shared__ __align__(32) _Float16 Plds[4][16][72];  // per-wave P tile, padded

  const int lane = threadIdx.x & 31;
  const int wid  = threadIdx.x >> 5;
  const int r    = lane & 15;
  const int hi   = lane >> 4;
  const int qb   = blockIdx.x;            // query block 0..31
  const int bh   = blockIdx.y;            // 0..31
  const int b    = bh >> 4;
  const int h    = bh & (HEADS - 1);
  const int q0   = qb * 64 + wid * 16;
  const float pc = pos_coeff[h];

  const _Float16* Qb = Qh + (size_t)bh * SEQ * HDIM;
  const _Float16* Kb = Kh + (size_t)bh * SEQ * HDIM;
  const _Float16* Vb = Vt + (size_t)bh * HDIM * SEQ;
  const float*    Bb = rel_bias + (size_t)b * SEQ * SEQ;
  const unsigned char* Mb = mask + (size_t)b * SEQ;

  // Q fragments stay resident (16 rows x 64 d = 2 K-steps)
  v16h qf[2];
#pragma unroll
  for (int kk = 0; kk < 2; ++kk)
    qf[kk] = load_a(Qb + (size_t)(q0 + r) * HDIM + kk * 32 + hi * 8);

  float m[8], l[8];
  v8f o[4] = {};
#pragma unroll
  for (int v = 0; v < 8; ++v) { m[v] = NEGF; l[v] = 0.f; }

  for (int kb = 0; kb < SEQ / 64; ++kb) {
    // sparsity: window blocks {qb-1,qb,qb+1}, global block 0; q-block 0 sees all
    if (qb != 0 && !(kb == 0 || (kb >= qb - 1 && kb <= qb + 1))) continue;
    const int kbase = kb * 64;

    // S = Q K^T
    v8f st[4] = {};
#pragma unroll
    for (int nt = 0; nt < 4; ++nt)
#pragma unroll
      for (int kk = 0; kk < 2; ++kk) {
        v16h kf = load_b(Kb + (size_t)(kbase + nt * 16 + r) * HDIM + kk * 32 + hi * 8);
        st[nt] = wmma_f16(qf[kk], kf, st[nt]);
      }

    // scale + rel-pos bias + sparse/global mask
#pragma unroll
    for (int nt = 0; nt < 4; ++nt) {
      const int kcol = kbase + nt * 16 + r;
      const unsigned char mk = Mb[kcol];
#pragma unroll
      for (int v = 0; v < 8; ++v) {
        const int q  = q0 + v + 8 * hi;
        const int dq = q - kcol;
        const int ad = dq < 0 ? -dq : dq;
        const bool valid = mk && (ad <= WIN || q == 0 || kcol == 0);
        const float x = st[nt][v] * 0.125f + pc * Bb[(size_t)q * SEQ + kcol];
        st[nt][v] = valid ? x : NEGF;
      }
    }

    // online softmax update (row stats: VGPR v <-> row v+8*hi; reduce across 16 lanes)
#pragma unroll
    for (int v = 0; v < 8; ++v) {
      float tl = fmaxf(fmaxf(st[0][v], st[1][v]), fmaxf(st[2][v], st[3][v]));
#pragma unroll
      for (int off = 1; off < 16; off <<= 1) tl = fmaxf(tl, __shfl_xor(tl, off, 32));
      const float mn = fmaxf(m[v], tl);
      const float sc = __expf(m[v] - mn);
      m[v] = mn;
      l[v] *= sc;
#pragma unroll
      for (int t = 0; t < 4; ++t) o[t][v] *= sc;
      float rs = 0.f;
#pragma unroll
      for (int nt = 0; nt < 4; ++nt) {
        const float sv = st[nt][v];
        const float p  = (sv > 0.5f * NEGF) ? __expf(sv - mn) : 0.f;  // guard all-masked tiles
        rs += p;
        Plds[wid][v + 8 * hi][nt * 16 + r] = (_Float16)p;             // C-layout -> LDS
      }
#pragma unroll
      for (int off = 1; off < 16; off <<= 1) rs += __shfl_xor(rs, off, 32);
      l[v] += rs;
    }

    asm volatile("s_wait_dscnt 0" ::: "memory");  // cross-lane LDS visibility (per-wave tile)

    // O += P V  (P re-read in A-layout; V is pre-transposed so B-frags are contiguous)
    v16h pf[2];
#pragma unroll
    for (int kk = 0; kk < 2; ++kk)
      pf[kk] = load_a(&Plds[wid][r][kk * 32 + hi * 8]);
#pragma unroll
    for (int dt = 0; dt < 4; ++dt)
#pragma unroll
      for (int kk = 0; kk < 2; ++kk) {
        v16h vf = load_b(Vb + (size_t)(dt * 16 + r) * SEQ + kbase + kk * 32 + hi * 8);
        o[dt] = wmma_f16(pf[kk], vf, o[dt]);
      }
  }

  // normalize and store as f16 [b, s, h*64+d] for the FC GEMM
#pragma unroll
  for (int dt = 0; dt < 4; ++dt)
#pragma unroll
    for (int v = 0; v < 8; ++v) {
      const int q = q0 + v + 8 * hi;
      const float val = o[dt][v] / l[v];
      attn_out[(size_t)(b * SEQ + q) * EMB + h * HDIM + dt * 16 + r] = (_Float16)val;
    }
}

// ---------------- launcher ----------------
extern "C" void kernel_launch(void* const* d_in, const int* in_sizes, int n_in,
                              void* d_out, int out_size, void* d_ws, size_t ws_size,
                              hipStream_t stream) {
  (void)in_sizes; (void)n_in; (void)out_size; (void)ws_size;
  const float* query    = (const float*)d_in[0];
  const float* key_     = (const float*)d_in[1];
  const float* value    = (const float*)d_in[2];
  const float* rel_bias = (const float*)d_in[3];
  const unsigned char* mask = (const unsigned char*)d_in[4];
  const float* wq_w = (const float*)d_in[5];
  const float* wq_b = (const float*)d_in[6];
  const float* wk_w = (const float*)d_in[7];
  const float* wk_b = (const float*)d_in[8];
  const float* wv_w = (const float*)d_in[9];
  const float* wv_b = (const float*)d_in[10];
  const float* fc_w = (const float*)d_in[11];
  const float* fc_b = (const float*)d_in[12];
  const float* pos_coeff = (const float*)d_in[13];

  char* ws = (char*)d_ws;
  const size_t HSZ = (size_t)BATCH * HEADS * SEQ * HDIM * sizeof(_Float16); // 8 MB
  const size_t XSZ = (size_t)NTOK * EMB * sizeof(_Float16);                 // 8 MB
  const size_t WSZ = (size_t)EMB * EMB * sizeof(_Float16);                  // 2 MB
  _Float16* Qh  = (_Float16*)(ws);             // 8 MB
  _Float16* Kh  = (_Float16*)(ws + 1 * HSZ);   // 8 MB
  _Float16* Vt  = (_Float16*)(ws + 2 * HSZ);   // 8 MB
  _Float16* Ao  = (_Float16*)(ws + 3 * HSZ);   // 8 MB
  _Float16* q16 = (_Float16*)(ws + 4 * HSZ);
  _Float16* k16 = (_Float16*)(q16 + NTOK * (size_t)EMB);
  _Float16* v16 = (_Float16*)(k16 + NTOK * (size_t)EMB);
  _Float16* wq16 = (_Float16*)(v16 + NTOK * (size_t)EMB);
  _Float16* wk16 = wq16 + (size_t)EMB * EMB;
  _Float16* wv16 = wk16 + (size_t)EMB * EMB;
  _Float16* wf16 = wv16 + (size_t)EMB * EMB;
  (void)XSZ; (void)WSZ;

  // one-time fp32->fp16 passes (memory-bound; keeps v_cvt out of the WMMA loops)
  const int NX = NTOK * EMB;      // 4M elems
  const int NW = EMB * EMB;       // 1M elems
  cvt_f16<<<NX / 2048, 256, 0, stream>>>(query, q16, NX);
  cvt_f16<<<NX / 2048, 256, 0, stream>>>(key_,  k16, NX);
  cvt_f16<<<NX / 2048, 256, 0, stream>>>(value, v16, NX);
  cvt_f16<<<NW / 2048, 256, 0, stream>>>(wq_w, wq16, NW);
  cvt_f16<<<NW / 2048, 256, 0, stream>>>(wk_w, wk16, NW);
  cvt_f16<<<NW / 2048, 256, 0, stream>>>(wv_w, wv16, NW);
  cvt_f16<<<NW / 2048, 256, 0, stream>>>(fc_w, wf16, NW);

  const dim3 ggrid(EMB / 256, NTOK / 64);  // (4, 64)
  const dim3 gblk(256);
  proj_gemm<MODE_Q><<<ggrid, gblk, 0, stream>>>(q16, wq16, wq_b, (void*)Qh);
  proj_gemm<MODE_K><<<ggrid, gblk, 0, stream>>>(k16, wk16, wk_b, (void*)Kh);
  proj_gemm<MODE_V><<<ggrid, gblk, 0, stream>>>(v16, wv16, wv_b, (void*)Vt);

  sparse_attn<<<dim3(SEQ / 64, BATCH * HEADS), dim3(128), 0, stream>>>(
      Qh, Kh, Vt, rel_bias, mask, pos_coeff, Ao);

  proj_gemm<MODE_FC><<<ggrid, gblk, 0, stream>>>(Ao, wf16, fc_b, d_out);
}
